// LlamaAttention_644245095212
// MI455X (gfx1250) — matrix operand order repaired and checked by reading, not compiled, not run on hardware
//
#include <hip/hip_runtime.h>
#include <hip/hip_bf16.h>

// ---------------------------------------------------------------------------
// Llama attention block for MI455X (gfx1250), wave32 + v_wmma_f32_16x16x32_bf16
// ---------------------------------------------------------------------------

typedef __attribute__((ext_vector_type(16))) __bf16 bf16x16;
typedef __attribute__((ext_vector_type(8)))  __bf16 bf16x8;
typedef __attribute__((ext_vector_type(8)))  float  f32x8;

union ABf { bf16x16 v; bf16x8 h[2]; };

constexpr int S_LEN = 2048;
constexpr int HID   = 4096;
constexpr int NH    = 32;
constexpr int NKV   = 8;
constexpr int HD    = 128;
constexpr int QW    = NH  * HD;   // 4096
constexpr int KW    = NKV * HD;   // 1024

// ---------------------------------------------------------------------------
// fp32 -> bf16 elementwise convert
// ---------------------------------------------------------------------------
__global__ void cvt_f32_bf16(const float* __restrict__ in,
                             __bf16* __restrict__ out, int n) {
  int i = blockIdx.x * blockDim.x + threadIdx.x;
  if (i < n) out[i] = (__bf16)in[i];
}

// ---------------------------------------------------------------------------
// C[M,N](f32, row-major) = A[M,K](bf16, row-major) @ W[N,K]^T (bf16, row-major)
// One wave computes a 32x64 tile (two A-frags share each B-frag -> 8 WMMA per
// 12 b128 loads). Block = 8 waves = 32 x 512 tile. grid = (N/512, M/32)
// ---------------------------------------------------------------------------
__global__ __launch_bounds__(256)
void gemm_bf16_nt(const __bf16* __restrict__ A, const __bf16* __restrict__ W,
                  float* __restrict__ C, int N, int K) {
  const int wave  = threadIdx.x >> 5;
  const int lane  = threadIdx.x & 31;
  const int mlane = lane & 15;          // A-row / B-col / C-col within tile
  const int half  = lane >> 4;          // which half of the wave
  const int m0 = blockIdx.y * 32;
  const int n0 = (blockIdx.x * 8 + wave) * 64;

  f32x8 acc[2][4] = {};
  const __bf16* Arow0 = A + (size_t)(m0 + mlane) * K;
  const __bf16* Arow1 = Arow0 + (size_t)16 * K;

  for (int k0 = 0; k0 < K; k0 += 32) {
    // A-frag 16x32: lane(m=mlane,h=half): elems 0-7 = K[8h..8h+7], 8-15 = K[16+8h..]
    ABf a0, a1;
    a0.h[0] = *(const bf16x8*)(Arow0 + k0 + 8 * half);
    a0.h[1] = *(const bf16x8*)(Arow0 + k0 + 16 + 8 * half);
    a1.h[0] = *(const bf16x8*)(Arow1 + k0 + 8 * half);
    a1.h[1] = *(const bf16x8*)(Arow1 + k0 + 16 + 8 * half);
#pragma unroll
    for (int t = 0; t < 4; ++t) {
      // B-frag 32x16: lane col n=mlane: elems 0-7 = K[8h..8h+7], 8-15 = K[16+8h..]
      const __bf16* wrow = W + (size_t)(n0 + t * 16 + mlane) * K + k0 + 8 * half;
      ABf b;
      b.h[0] = *(const bf16x8*)(wrow);
      b.h[1] = *(const bf16x8*)(wrow + 16);
      acc[0][t] = __builtin_amdgcn_wmma_f32_16x16x32_bf16(
          false, a0.v, false, b.v, (short)0, acc[0][t], false, false);
      acc[1][t] = __builtin_amdgcn_wmma_f32_16x16x32_bf16(
          false, a1.v, false, b.v, (short)0, acc[1][t], false, false);
    }
  }
  // C/D layout: VGPR r -> row (m0 + 16*g) + r + 8*half, col n0 + t*16 + mlane
#pragma unroll
  for (int g = 0; g < 2; ++g)
#pragma unroll
    for (int t = 0; t < 4; ++t)
#pragma unroll
      for (int r = 0; r < 8; ++r)
        C[(size_t)(m0 + 16 * g + r + 8 * half) * N + n0 + t * 16 + mlane] =
            acc[g][t][r];
}

// ---------------------------------------------------------------------------
// RoPE + bf16 convert (optionally folds the 1/sqrt(d) softmax scale into Q)
// in/out layout [S, nheads*128]; one thread per (s, head, d<64) pair
// ---------------------------------------------------------------------------
__global__ void rope_cvt(const float* __restrict__ inF,
                         const int* __restrict__ pos_ids,
                         __bf16* __restrict__ outB, int nheads, float outScale) {
  int idx = blockIdx.x * blockDim.x + threadIdx.x;
  int total = S_LEN * nheads * 64;
  if (idx >= total) return;
  int d = idx & 63;
  int h = (idx >> 6) % nheads;
  int s = idx / (64 * nheads);
  int W = nheads * HD;
  float pos  = (float)pos_ids[s];
  // inv_freq = 10000^(-2d/128) = exp(-d/64 * ln(10000))
  float ang  = pos * __expf(-(float)d * 0.015625f * 9.2103403719761836f);
  float c = __cosf(ang), sn = __sinf(ang);
  size_t base = (size_t)s * W + h * HD;
  float x1 = inF[base + d], x2 = inF[base + 64 + d];
  outB[base + d]      = (__bf16)((x1 * c - x2 * sn) * outScale);
  outB[base + 64 + d] = (__bf16)((x2 * c + x1 * sn) * outScale);
}

// ---------------------------------------------------------------------------
// V: fp32 [S, KW] -> bf16 transposed [KW, S] (so PV B-frags are contiguous)
// ---------------------------------------------------------------------------
__global__ void transpose_cvt_v(const float* __restrict__ Vf,
                                __bf16* __restrict__ Vt) {
  int idx = blockIdx.x * blockDim.x + threadIdx.x;
  if (idx >= S_LEN * KW) return;
  int d = idx & (KW - 1);
  int s = idx / KW;
  Vt[(size_t)d * S_LEN + s] = (__bf16)Vf[(size_t)s * KW + d];
}

// ---------------------------------------------------------------------------
// Flash attention: one wave = one (head, 16-query tile). 8 waves / block.
// Qb [S, QW] pre-scaled by 1/sqrt(d); Kb [S, KW]; Vt [KW, S]; Ob [S, QW] bf16
// ---------------------------------------------------------------------------
__global__ __launch_bounds__(256)
void attn_fa(const __bf16* __restrict__ Qb, const __bf16* __restrict__ Kb,
             const __bf16* __restrict__ Vt, __bf16* __restrict__ Ob) {
  __shared__ __bf16 ldsP[8][16 * 32];       // 1KB P tile per wave
  const int wave  = threadIdx.x >> 5;
  const int lane  = threadIdx.x & 31;
  const int mlane = lane & 15;
  const int half  = lane >> 4;
  const int gid   = blockIdx.x * 8 + wave;  // 0..4095
  const int head  = gid >> 7;               // / (S/16)
  const int q0    = (gid & 127) * 16;
  const int kvh   = head >> 2;              // GQA: 4 Q heads per KV head
  __bf16* myP = ldsP[wave];

  // Q tile 16x128 as four 16x32 A-fragments, kept in VGPRs
  ABf qf[4];
  const __bf16* qrow = Qb + (size_t)(q0 + mlane) * QW + head * HD;
#pragma unroll
  for (int c = 0; c < 4; ++c) {
    qf[c].h[0] = *(const bf16x8*)(qrow + c * 32 + 8 * half);
    qf[c].h[1] = *(const bf16x8*)(qrow + c * 32 + 16 + 8 * half);
  }

  f32x8 oacc[8] = {};
  float mrow[8], lrow[8];
#pragma unroll
  for (int r = 0; r < 8; ++r) { mrow[r] = -1e30f; lrow[r] = 0.f; }

  for (int kb = 0; kb <= q0 + 15; kb += 32) {
    // ---- scores: 16 queries x 32 keys = two 16x16 C tiles, K-dim = 128 ----
    f32x8 sc[2] = {};
#pragma unroll
    for (int t = 0; t < 2; ++t)
#pragma unroll
      for (int c = 0; c < 4; ++c) {
        const __bf16* krow = Kb + (size_t)(kb + t * 16 + mlane) * KW +
                             kvh * HD + c * 32 + 8 * half;
        ABf b;
        b.h[0] = *(const bf16x8*)(krow);
        b.h[1] = *(const bf16x8*)(krow + 16);
        sc[t] = __builtin_amdgcn_wmma_f32_16x16x32_bf16(
            false, qf[c].v, false, b.v, (short)0, sc[t], false, false);
      }

    // ---- causal mask + online softmax (per-row, rows r+8*half) ----
#pragma unroll
    for (int r = 0; r < 8; ++r) {
      const int qg = q0 + r + 8 * half;
#pragma unroll
      for (int t = 0; t < 2; ++t)
        if (kb + t * 16 + mlane > qg) sc[t][r] = -1e30f;

      float mx = fmaxf(sc[0][r], sc[1][r]);
#pragma unroll
      for (int off = 1; off < 16; off <<= 1)
        mx = fmaxf(mx, __shfl_xor(mx, off, 32));   // reduce over 16-lane group
      float mn = fmaxf(mrow[r], mx);
      float p0 = __expf(sc[0][r] - mn);
      float p1 = __expf(sc[1][r] - mn);
      sc[0][r] = p0; sc[1][r] = p1;
      float ls = p0 + p1;
#pragma unroll
      for (int off = 1; off < 16; off <<= 1)
        ls += __shfl_xor(ls, off, 32);
      float corr = __expf(mrow[r] - mn);
      lrow[r] = lrow[r] * corr + ls;
      mrow[r] = mn;
#pragma unroll
      for (int nt = 0; nt < 8; ++nt) oacc[nt][r] *= corr;
    }

    // ---- re-layout P: C-layout regs -> LDS [16 rows][32 keys] -> A-frag ----
#pragma unroll
    for (int r = 0; r < 8; ++r) {
      myP[(r + 8 * half) * 32 + mlane]      = (__bf16)sc[0][r];
      myP[(r + 8 * half) * 32 + 16 + mlane] = (__bf16)sc[1][r];
    }
    // wave-private LDS region; DS ops are in-order within a wave
    ABf pf;
    pf.h[0] = *(const bf16x8*)(myP + mlane * 32 + 8 * half);
    pf.h[1] = *(const bf16x8*)(myP + mlane * 32 + 16 + 8 * half);

    // ---- O += P(16x32) x V(32x16) over the 8 head-dim tiles ----
#pragma unroll
    for (int nt = 0; nt < 8; ++nt) {
      const __bf16* vrow = Vt + (size_t)(kvh * HD + nt * 16 + mlane) * S_LEN +
                           kb + 8 * half;
      ABf b;
      b.h[0] = *(const bf16x8*)(vrow);
      b.h[1] = *(const bf16x8*)(vrow + 16);
      oacc[nt] = __builtin_amdgcn_wmma_f32_16x16x32_bf16(
          false, pf.v, false, b.v, (short)0, oacc[nt], false, false);
    }
  }

  // ---- finalize: O / l, store bf16 [S, QW] ----
#pragma unroll
  for (int r = 0; r < 8; ++r) {
    float inv = 1.0f / lrow[r];
#pragma unroll
    for (int nt = 0; nt < 8; ++nt)
      Ob[(size_t)(q0 + r + 8 * half) * QW + head * HD + nt * 16 + mlane] =
          (__bf16)(oacc[nt][r] * inv);
  }
}

// ---------------------------------------------------------------------------
// Host launcher. Workspace use ~185 MB.
// ---------------------------------------------------------------------------
extern "C" void kernel_launch(void* const* d_in, const int* in_sizes, int n_in,
                              void* d_out, int out_size, void* d_ws, size_t ws_size,
                              hipStream_t stream) {
  (void)in_sizes; (void)n_in; (void)out_size; (void)ws_size;
  const float* X  = (const float*)d_in[0];
  const int*   pos = (const int*)d_in[1];
  const float* Wq = (const float*)d_in[2];
  const float* Wk = (const float*)d_in[3];
  const float* Wv = (const float*)d_in[4];
  const float* Wo = (const float*)d_in[5];
  float* out = (float*)d_out;

  char* ws = (char*)d_ws;
  size_t off = 0;
  auto alloc = [&](size_t bytes) {
    char* p = ws + off;
    off += (bytes + 255) & ~(size_t)255;
    return p;
  };
  __bf16* Xb  = (__bf16*)alloc((size_t)S_LEN * HID * 2);
  __bf16* Wqb = (__bf16*)alloc((size_t)QW * HID * 2);
  __bf16* Wkb = (__bf16*)alloc((size_t)KW * HID * 2);
  __bf16* Wvb = (__bf16*)alloc((size_t)KW * HID * 2);
  __bf16* Wob = (__bf16*)alloc((size_t)HID * QW * 2);
  float*  Qf  = (float*) alloc((size_t)S_LEN * QW * 4);
  float*  Kf  = (float*) alloc((size_t)S_LEN * KW * 4);
  float*  Vf  = (float*) alloc((size_t)S_LEN * KW * 4);
  __bf16* Qb  = (__bf16*)alloc((size_t)S_LEN * QW * 2);
  __bf16* Kb  = (__bf16*)alloc((size_t)S_LEN * KW * 2);
  __bf16* Vt  = (__bf16*)alloc((size_t)KW * S_LEN * 2);
  __bf16* Ab  = (__bf16*)alloc((size_t)S_LEN * QW * 2);

  auto cvt = [&](const float* in, __bf16* o, int n) {
    cvt_f32_bf16<<<dim3((n + 255) / 256), dim3(256), 0, stream>>>(in, o, n);
  };
  cvt(X,  Xb,  S_LEN * HID);
  cvt(Wq, Wqb, QW * HID);
  cvt(Wk, Wkb, KW * HID);
  cvt(Wv, Wvb, KW * HID);
  cvt(Wo, Wob, HID * QW);

  // Projections: [S,H] x W^T  (grid = N/512 x M/32)
  gemm_bf16_nt<<<dim3(QW / 512, S_LEN / 32), dim3(256), 0, stream>>>(Xb, Wqb, Qf, QW, HID);
  gemm_bf16_nt<<<dim3(KW / 512, S_LEN / 32), dim3(256), 0, stream>>>(Xb, Wkb, Kf, KW, HID);
  gemm_bf16_nt<<<dim3(KW / 512, S_LEN / 32), dim3(256), 0, stream>>>(Xb, Wvb, Vf, KW, HID);

  // RoPE (+ 1/sqrt(128) fused into Q) and V transpose-convert
  const float kScale = 0.08838834764831845f;  // 1/sqrt(128)
  rope_cvt<<<dim3((S_LEN * NH  * 64) / 256), dim3(256), 0, stream>>>(Qf, pos, Qb, NH,  kScale);
  rope_cvt<<<dim3((S_LEN * NKV * 64) / 256), dim3(256), 0, stream>>>(Kf, pos, Kb, NKV, 1.0f);
  transpose_cvt_v<<<dim3((S_LEN * KW) / 256), dim3(256), 0, stream>>>(Vf, Vt);

  // Flash attention: 32 heads x 128 q-tiles = 4096 waves, 8 waves/block
  attn_fa<<<dim3((NH * (S_LEN / 16)) / 8), dim3(256), 0, stream>>>(Qb, Kb, Vt, Ab);

  // Output projection -> fp32 result
  gemm_bf16_nt<<<dim3(HID / 512, S_LEN / 32), dim3(256), 0, stream>>>(Ab, Wob, out, HID, QW);
}
